// DirectMHPInfer_43937515438316
// MI455X (gfx1250) — 8-bit matrix operand order repaired and checked by reading, compiled, not run on hardware
//
#include <hip/hip_runtime.h>
#include <stdint.h>

#define BATCH   64
#define NPRED   100800
#define NCH     9
#define KDET    512
#define CONF_T  0.7f
#define IOU_T   0.45f
#define NBINS   2048
#define CAP     1024

typedef int v8i __attribute__((ext_vector_type(8)));

__global__ __launch_bounds__(1024)
void k_hist(const float* __restrict__ pred, float* __restrict__ cutVal,
            unsigned* __restrict__ cnt) {
    __shared__ unsigned hist[NBINS];
    __shared__ int cutBin;
    const int b   = blockIdx.x;
    const int tid = threadIdx.x;

    for (int i = tid; i < NBINS; i += 1024) hist[i] = 0u;
    if (tid == 0) cutBin = 0;
    __syncthreads();

    const float* base  = pred + (size_t)b * NPRED * NCH;
    const float  scale = (float)NBINS / (1.0f - CONF_T);
    for (int n = tid; n < NPRED; n += 1024) {
        const float* r = base + (size_t)n * NCH;
        __builtin_prefetch(base + (size_t)(n + 1024) * NCH, 0, 1);
        float c = r[4] * r[5];
        if (c > CONF_T) {
            int bin = (int)((c - CONF_T) * scale);
            bin = bin < 0 ? 0 : (bin > NBINS - 1 ? NBINS - 1 : bin);
            atomicAdd(&hist[bin], 1u);
        }
    }
    __syncthreads();

    for (int s = 1; s < NBINS; s <<= 1) {
        const int i0 = tid, i1 = tid + 1024;
        unsigned t0 = (i0 + s < NBINS) ? hist[i0 + s] : 0u;
        unsigned t1 = (i1 + s < NBINS) ? hist[i1 + s] : 0u;
        __syncthreads();
        hist[i0] += t0;
        hist[i1] += t1;
        __syncthreads();
    }

    for (int i = tid; i < NBINS; i += 1024) {
        unsigned here  = hist[i];
        unsigned above = (i + 1 < NBINS) ? hist[i + 1] : 0u;
        if (here >= KDET && above < KDET) cutBin = i;
    }
    __syncthreads();
    if (tid == 0) {
        cutVal[b] = CONF_T + (float)cutBin * ((1.0f - CONF_T) / (float)NBINS);
        cnt[b]    = 0u;
    }
}

__global__ __launch_bounds__(512)
void k_collect(const float* __restrict__ pred, const float* __restrict__ cutVal,
               unsigned* __restrict__ cnt, float2* __restrict__ cand) {
    const int b  = blockIdx.x;
    const float cv = cutVal[b];
    const float* base = pred + (size_t)b * NPRED * NCH;
    const int start  = blockIdx.y * blockDim.x + threadIdx.x;
    const int stride = gridDim.y * blockDim.x;
    for (int n = start; n < NPRED; n += stride) {
        const float* r = base + (size_t)n * NCH;
        __builtin_prefetch(base + (size_t)(n + stride) * NCH, 0, 1);
        float c = r[4] * r[5];
        if (c > cv) {
            unsigned pos = atomicAdd(&cnt[b], 1u);
            if (pos < CAP) {
                float2 e;
                e.x = c;
                e.y = __int_as_float(n);
                cand[(size_t)b * CAP + pos] = e;
            }
        }
    }
}

__device__ __forceinline__ float iou_lds(const float* x1, const float* y1,
                                         const float* x2, const float* y2,
                                         int i, int j) {
    float lx = fmaxf(x1[i], x1[j]);
    float ly = fmaxf(y1[i], y1[j]);
    float rx = fminf(x2[i], x2[j]);
    float ry = fminf(y2[i], y2[j]);
    float w  = fmaxf(rx - lx, 0.0f);
    float h  = fmaxf(ry - ly, 0.0f);
    float inter = w * h;
    float ai = (x2[i] - x1[i]) * (y2[i] - y1[i]);
    float aj = (x2[j] - x1[j]) * (y2[j] - y1[j]);
    return inter / (ai + aj - inter + 1e-7f);
}

__global__ __launch_bounds__(512)
void k_nms(const float* __restrict__ pred, const float2* __restrict__ cand,
           const unsigned* __restrict__ cnt, float* __restrict__ out) {
    __shared__ float key[CAP];
    __shared__ int   idxA[CAP];
    __shared__ float bx1[KDET], by1[KDET], bx2[KDET], by2[KDET];
    __shared__ float sc[KDET];
    __shared__ float an0[KDET], an1[KDET], an2[KDET];
    __shared__ unsigned keepW[KDET / 4];
    __shared__ unsigned SbufW[64 * KDET / 4];
    __shared__ int  deadCnt[64];
    __shared__ unsigned char alive[64];

    const int b   = blockIdx.x;
    const int tid = threadIdx.x;
    const int lane = tid & 31;
    const int wave = tid >> 5;

    unsigned cb = cnt[b];
    int cntb = (cb < CAP) ? (int)cb : CAP;

    for (int i = tid; i < CAP; i += 512) {
        if (i < cntb) {
            float2 e = cand[(size_t)b * CAP + i];
            key[i]  = e.x;
            idxA[i] = __float_as_int(e.y);
        } else {
            key[i]  = -1.0f;
            idxA[i] = 0;
        }
    }
    __syncthreads();

    for (int kk = 2; kk <= CAP; kk <<= 1) {
        for (int jj = kk >> 1; jj > 0; jj >>= 1) {
            for (int i = tid; i < CAP; i += 512) {
                int ixj = i ^ jj;
                if (ixj > i) {
                    bool desc = ((i & kk) == 0);
                    float ki = key[i], kx = key[ixj];
                    bool sw = desc ? (ki < kx) : (ki > kx);
                    if (sw) {
                        key[i] = kx; key[ixj] = ki;
                        int ti = idxA[i]; idxA[i] = idxA[ixj]; idxA[ixj] = ti;
                    }
                }
            }
            __syncthreads();
        }
    }

    {
        const int k = tid;
        int src = idxA[k];
        const float* r = pred + ((size_t)b * NPRED + (size_t)src) * NCH;
        float cx = r[0], cy = r[1], w = r[2], h = r[3];
        bx1[k] = cx - 0.5f * w;
        by1[k] = cy - 0.5f * h;
        bx2[k] = cx + 0.5f * w;
        by2[k] = cy + 0.5f * h;
        an0[k] = r[6]; an1[k] = r[7]; an2[k] = r[8];
        float s = key[k];
        sc[k] = s;
        ((unsigned char*)keepW)[k] = (s > CONF_T) ? 1u : 0u;
    }
    __syncthreads();

    for (int t = 0; t < 8; ++t) {
        __syncthreads();
        const int prior = 64 * t;
        for (int e = tid; e < prior * 64; e += 512) {
            int i = e % prior;
            int j = e / prior;
            unsigned char s =
                (iou_lds(bx1, by1, bx2, by2, i, prior + j) > IOU_T) ? 1u : 0u;
            ((unsigned char*)SbufW)[j * KDET + i] = s;
        }
        __syncthreads();

        if (t > 0 && wave < 4) {
            const int g   = wave;
            const int lh2 = (lane >= 16) ? 2 : 0;
            const int lb  = (lane >= 16) ? 16 : 0;
            const int j   = g * 16 + (lane & 15);
            v8i acc = (v8i){0, 0, 0, 0, 0, 0, 0, 0};
            for (int c = 0; c < t; ++c) {
                v8i a, bf;
#pragma unroll
                for (int v = 0; v < 8; ++v) {
                    a[v] = (int)keepW[c * 16 + (v & 3) + ((v >> 2) << 3) + 2 * lh2];
                    int byteOff = j * KDET + c * 64 + lb + ((v & 3) << 2) +
                                  ((v >= 4) ? 32 : 0);
                    bf[v] = (int)SbufW[byteOff >> 2];
                }
                acc = __builtin_amdgcn_wmma_i32_16x16x64_iu8(
                    false, a, false, bf, acc, false, false);
            }
            if (lane < 16) deadCnt[g * 16 + lane] = acc[0];
        }
        __syncthreads();

        if (tid < 64) {
            int d = (t > 0) ? deadCnt[tid] : 0;
            alive[tid] =
                (((unsigned char*)keepW)[prior + tid] != 0u) && (d == 0);
        }
        __syncthreads();
        for (int ii = 0; ii < 63; ++ii) {
            if (tid < 64 && tid > ii && alive[ii] && alive[tid]) {
                if (iou_lds(bx1, by1, bx2, by2, prior + ii, prior + tid) > IOU_T)
                    alive[tid] = 0u;
            }
            __syncthreads();
        }
        if (tid < 64) ((unsigned char*)keepW)[prior + tid] = alive[tid];
        __syncthreads();
    }

    {
        const int k = tid;
        float m = (((unsigned char*)keepW)[k] != 0u) ? 1.0f : 0.0f;
        float* o = out + ((size_t)b * KDET + k) * NCH;
        o[0] = bx1[k] * m;
        o[1] = by1[k] * m;
        o[2] = bx2[k] * m;
        o[3] = by2[k] * m;
        o[4] = sc[k] * m;
        o[5] = 0.0f;
        o[6] = an0[k] * m;
        o[7] = an1[k] * m;
        o[8] = an2[k] * m;
    }
}

extern "C" void kernel_launch(void* const* d_in, const int* in_sizes, int n_in,
                              void* d_out, int out_size, void* d_ws, size_t ws_size,
                              hipStream_t stream) {
    (void)in_sizes; (void)n_in; (void)out_size; (void)ws_size;
    const float* pred = (const float*)d_in[0];
    float* out = (float*)d_out;

    char* ws = (char*)d_ws;
    float*    cutVal = (float*)ws;
    unsigned* cnt    = (unsigned*)(ws + 256);
    float2*   cand   = (float2*)(ws + 512);

    k_hist<<<BATCH, 1024, 0, stream>>>(pred, cutVal, cnt);
    k_collect<<<dim3(BATCH, 8), 512, 0, stream>>>(pred, cutVal, cnt, cand);
    k_nms<<<BATCH, 512, 0, stream>>>(pred, cand, cnt, out);
}
